// DN4_12266426597442
// MI455X (gfx1250) — matrix operand order repaired and checked by reading, not compiled, not run on hardware
//
#include <hip/hip_runtime.h>

typedef __attribute__((ext_vector_type(16))) _Float16 v16h;
typedef __attribute__((ext_vector_type(8)))  _Float16 v8h;
typedef __attribute__((ext_vector_type(8)))  float    v8f;

#define WAYS        5
#define SHOTS       5
#define QUERIES     30
#define CCH         128
#define HW          441           // 21*21
#define NS          25            // WAYS*SHOTS
#define NIMG        55            // NS + QUERIES
#define NCLS        (SHOTS*HW)    // 2205 descriptors per class bank
#define MTILES      28            // ceil(441/16)
#define NTILES      138           // ceil(2205/16)
#define WPB         4             // waves per block in the GEMM kernel
#define NEGINF      (-3.0e38f)

// Branchless insert of v into sorted triple (t0 >= t1 >= t2): 5 min/max ops
// (compiler fuses these into v_minmax_num / dual-issue pairs).
__device__ __forceinline__ void top3_insert(float& t0, float& t1, float& t2, float v) {
  const float n0 = fmaxf(t0, v);
  const float l0 = fminf(t0, v);
  const float n1 = fmaxf(t1, l0);
  const float l1 = fminf(t1, l0);
  const float n2 = fmaxf(t2, l1);
  t0 = n0; t1 = n1; t2 = n2;
}

__device__ __forceinline__ v8f wmma_f16(const v16h& a, const v16h& b, const v8f& c) {
  return __builtin_amdgcn_wmma_f32_16x16x32_f16(false, a, false, b, (short)0, c,
                                                false, false);
}

// ---------------------------------------------------------------------------
// Kernel 1: one-hot labels -> fwdMap[img] = class*SHOTS + rank-within-class
// (reproduces stable-argsort grouping for any one-hot assignment).
// Runs BEFORE normalization so support descriptors can be scattered directly
// into class-ordered banks.
// ---------------------------------------------------------------------------
__global__ void dn4_classmap(const float* __restrict__ labels,
                             int* __restrict__ fwdMap) {
  const int s = threadIdx.x;
  if (s >= NS) return;
  auto cls_of = [&](int si) {
    int best = 0; float bv = labels[si * WAYS];
    for (int w = 1; w < WAYS; ++w) {
      const float v = labels[si * WAYS + w];
      if (v > bv) { bv = v; best = w; }
    }
    return best;
  };
  const int c = cls_of(s);
  int rank = 0;
  for (int t = 0; t < s; ++t) if (cls_of(t) == c) ++rank;
  fwdMap[s] = c * SHOTS + rank;
}

// ---------------------------------------------------------------------------
// Kernel 2: L2-normalize each [C]-descriptor, convert to f16, and write to
//   rows [0, NS*HW)      : class-ordered support banks  [way][shot][hw][C]
//   rows [NS*HW, NIMG*HW): query descriptors            [q][hw][C]
// One wave32 per descriptor; 8 descriptors per 256-thread block.
// ---------------------------------------------------------------------------
__global__ void dn4_normalize(const float* __restrict__ sup,
                              const float* __restrict__ qry,
                              const int* __restrict__ fwdMap,
                              _Float16* __restrict__ zf16, int ndesc) {
  const int wave = threadIdx.x >> 5;
  const int lane = threadIdx.x & 31;
  const int d = blockIdx.x * (blockDim.x >> 5) + wave;
  if (d >= ndesc) return;
  const int img = d / HW;
  const int hw  = d - img * HW;
  const bool isSup = (img < NS);
  const float* src = isSup ? (sup + (size_t)img * CCH * HW)
                           : (qry + (size_t)(img - NS) * CCH * HW);
  float vals[4];
  float ss = 0.f;
#pragma unroll
  for (int i = 0; i < 4; ++i) {
    const int c = i * 32 + lane;
    const float v = src[(size_t)c * HW + hw];
    vals[i] = v;
    ss += v * v;
  }
#pragma unroll
  for (int m = 16; m >= 1; m >>= 1) ss += __shfl_xor(ss, m, 32);
  const float inv = 1.f / fmaxf(sqrtf(ss), 1e-12f);
  const int drow = isSup ? (fwdMap[img] * HW + hw) : d;   // scatter support by class
  _Float16* dst = zf16 + (size_t)drow * CCH;
#pragma unroll
  for (int i = 0; i < 4; ++i) dst[i * 32 + lane] = (_Float16)(vals[i] * inv);
}

// ---------------------------------------------------------------------------
// Kernel 3: per (q, w, mtile): 16x2205 similarity strip via WMMA f16->f32,
// two n-tiles per iteration (4 independent WMMA chains), fused branchless
// per-row top-3, cross-wave merge in LDS. EXEC stays full everywhere.
// ---------------------------------------------------------------------------
__global__ __launch_bounds__(WPB * 32, 1)
void dn4_sim_topk(const _Float16* __restrict__ zf16,   // see layout above
                  float* __restrict__ partials) {
  __shared__ float lds_top[WPB * 32 * 24];
  __shared__ float lds_row[16];

  const int mt   = blockIdx.x;
  const int w    = blockIdx.y;
  const int q    = blockIdx.z;
  const int wave = threadIdx.x >> 5;
  const int lane = threadIdx.x & 31;
  const int hh   = lane >> 4;       // half-wave id
  const int sub  = lane & 15;       // row (A) / column (B) within the tile

  const _Float16* zq = zf16 + (size_t)NS * HW * CCH;

  // ---- A fragments: 16 query rows x K=128, kept in registers for the n loop.
  // A layout (matches B): half 0 holds K 0..15 contiguous, half 1 holds K 16..31.
  v16h afrag[4];
  {
    const int m = min(mt * 16 + sub, HW - 1);   // padded rows excluded later
    const _Float16* rowp = zq + ((size_t)q * HW + m) * CCH + hh * 16;
#pragma unroll
    for (int kk = 0; kk < 4; ++kk) {
      const v8h lo = *(const v8h*)(rowp + kk * 32);
      const v8h hi = *(const v8h*)(rowp + kk * 32 + 8);
      afrag[kk] = __builtin_shufflevector(lo, hi, 0,1,2,3,4,5,6,7,8,9,10,11,12,13,14,15);
    }
  }

  // Per-lane running top-3 for 8 row-slots.
  float t0[8], t1[8], t2[8];
#pragma unroll
  for (int v = 0; v < 8; ++v) { t0[v] = NEGINF; t1[v] = NEGINF; t2[v] = NEGINF; }

  // B pointer: class bank is contiguous, column gn at byte offset gn*256.
  // B layout (matches A): half 0 holds K 0..15 contiguous, half 1 holds K 16..31.
  const _Float16* bp = zf16 + ((size_t)w * NCLS + wave * 32 + sub) * CCH + hh * 16;
  int gn = wave * 32 + sub;                        // column id for tile pair lead

  for (int base = wave * 2; base < NTILES; base += WPB * 2) {
    const _Float16* c0 = bp;
    const _Float16* c1 = bp + 16 * CCH;
    v16h b0[4], b1[4];
#pragma unroll
    for (int kk = 0; kk < 4; ++kk) b0[kk] = *(const v16h*)(c0 + kk * 32);
#pragma unroll
    for (int kk = 0; kk < 4; ++kk) b1[kk] = *(const v16h*)(c1 + kk * 32);

    // Four independent accumulation chains (2 tiles x K even/odd) for ILP.
    v8f p0 = {}, p1 = {}, s0 = {}, s1 = {};
    p0 = wmma_f16(afrag[0], b0[0], p0);
    p1 = wmma_f16(afrag[1], b0[1], p1);
    s0 = wmma_f16(afrag[0], b1[0], s0);
    s1 = wmma_f16(afrag[1], b1[1], s1);
    p0 = wmma_f16(afrag[2], b0[2], p0);
    p1 = wmma_f16(afrag[3], b0[3], p1);
    s0 = wmma_f16(afrag[2], b1[2], s0);
    s1 = wmma_f16(afrag[3], b1[3], s1);

    const bool v0 = (gn < NCLS);
    const bool v1 = (gn + 16 < NCLS);
#pragma unroll
    for (int v = 0; v < 8; ++v) {
      const float val0 = v0 ? (p0[v] + p1[v]) : NEGINF;
      top3_insert(t0[v], t1[v], t2[v], val0);
      const float val1 = v1 ? (s0[v] + s1[v]) : NEGINF;
      top3_insert(t0[v], t1[v], t2[v], val1);
    }

    gn += WPB * 32;
    bp += (size_t)(WPB * 32) * CCH;
  }

  // ---- dump per-lane top-3 candidates and merge per row.
  {
    const int bofs = (wave * 32 + lane) * 24;
#pragma unroll
    for (int v = 0; v < 8; ++v) {
      lds_top[bofs + v * 3 + 0] = t0[v];
      lds_top[bofs + v * 3 + 1] = t1[v];
      lds_top[bofs + v * 3 + 2] = t2[v];
    }
  }
  __syncthreads();

  if (threadIdx.x < 16) {
    const int M = threadIdx.x;            // row within the tile
    const int v = M & 7;                  // VGPR slot
    const int lstart = (M >> 3) * 16;     // lanes holding this row
    float a0 = NEGINF, a1 = NEGINF, a2 = NEGINF;
    for (int wv = 0; wv < WPB; ++wv) {
      for (int L = lstart; L < lstart + 16; ++L) {
        const int bofs = (wv * 32 + L) * 24 + v * 3;
#pragma unroll
        for (int j = 0; j < 3; ++j)
          top3_insert(a0, a1, a2, lds_top[bofs + j]);
      }
    }
    const int m = mt * 16 + M;
    lds_row[M] = (m < HW) ? (a0 + a1 + a2) : 0.f;
  }
  __syncthreads();

  if (threadIdx.x == 0) {
    float s = 0.f;
#pragma unroll
    for (int i = 0; i < 16; ++i) s += lds_row[i];
    partials[((q * WAYS + w) * MTILES) + mt] = s;
  }
}

// ---------------------------------------------------------------------------
// Kernel 4: deterministic reduction of the 28 m-tile partials + mean scale.
// ---------------------------------------------------------------------------
__global__ void dn4_finalize(const float* __restrict__ partials,
                             float* __restrict__ out) {
  const int i = blockIdx.x * blockDim.x + threadIdx.x;
  if (i >= QUERIES * WAYS) return;
  float s = 0.f;
  for (int t = 0; t < MTILES; ++t) s += partials[i * MTILES + t];
  out[i] = s * (1.f / (float)(HW * 3));
}

// ---------------------------------------------------------------------------
extern "C" void kernel_launch(void* const* d_in, const int* in_sizes, int n_in,
                              void* d_out, int out_size, void* d_ws, size_t ws_size,
                              hipStream_t stream) {
  (void)in_sizes; (void)n_in; (void)out_size; (void)ws_size;
  const float* sup = (const float*)d_in[0];
  const float* lab = (const float*)d_in[1];
  const float* qry = (const float*)d_in[2];
  float* out = (float*)d_out;

  char* ws = (char*)d_ws;
  _Float16* zf16 = (_Float16*)ws;                         // 55*441*128 f16 (~5.9 MB)
  size_t off = ((size_t)NIMG * HW * CCH * sizeof(_Float16) + 255) & ~(size_t)255;
  int* fwdMap = (int*)(ws + off);
  off += 256;
  float* partials = (float*)(ws + off);                   // 30*5*28 floats

  const int ndesc = NIMG * HW;
  dn4_classmap<<<1, 32, 0, stream>>>(lab, fwdMap);
  dn4_normalize<<<(ndesc + 7) / 8, 256, 0, stream>>>(sup, qry, fwdMap, zf16, ndesc);
  dim3 grid(MTILES, WAYS, QUERIES);
  dn4_sim_topk<<<grid, WPB * 32, 0, stream>>>(zf16, partials);
  dn4_finalize<<<1, 256, 0, stream>>>(partials, out);
}
